// MultiheadAttentionBlock_9912784519946
// MI455X (gfx1250) — matrix operand order repaired and checked by reading, not compiled, not run on hardware
//
#include <hip/hip_runtime.h>

typedef __bf16 bf16_t;
typedef bf16_t v16bf __attribute__((ext_vector_type(16)));
typedef float  v8f   __attribute__((ext_vector_type(8)));

#define D_MODEL 1024
#define NHEAD   16
#define DK      64
#define SEQ     2048
#define BATCH   4
#define MROWS   (BATCH * SEQ)

union FragU { uint4 u[2]; v16bf v; };

__device__ __forceinline__ v8f v8f_zero() {
  v8f z;
#pragma unroll
  for (int i = 0; i < 8; ++i) z[i] = 0.0f;
  return z;
}

__device__ __forceinline__ v8f wmma_acc(v16bf a, v16bf b, v8f c) {
  // D = A(16x32 bf16) * B(32x16 bf16) + C(16x16 f32)
  return __builtin_amdgcn_wmma_f32_16x16x32_bf16(false, a, false, b, (short)0, c,
                                                 false, false);
}

// A fragment (16x32): lane&15 = row; lanes 0-15 hold K {klo..klo+7, klo+16..klo+23}
// with klo=0; lanes 16-31 same with klo=8. Two contiguous 16B loads per lane.
__device__ __forceinline__ v16bf load_frag_a(const bf16_t* __restrict__ A,
                                             int row0, int k, int ld) {
  const int lane = threadIdx.x & 31;
  const int m    = lane & 15;
  const int klo  = (lane >> 4) << 3;  // 0 or 8
  const bf16_t* p = A + (size_t)(row0 + m) * ld + k + klo;
  FragU f;
  f.u[0] = *(const uint4*)(p);        // K = k+klo .. k+klo+7
  f.u[1] = *(const uint4*)(p + 16);   // K = k+klo+16 .. k+klo+23
  return f.v;
}

// B fragment (32x16): lane&15 = col; lanes 0-15 hold K {0..7, 16..23}, lanes 16-31 K {8..15, 24..31}.
// Bm stored with K contiguous per column: Bm[col*ld + k]. One 32B (2x16B) load.
__device__ __forceinline__ v16bf load_frag_b(const bf16_t* __restrict__ Bm,
                                             int col0, int k, int ld) {
  const int lane = threadIdx.x & 31;
  const int n    = lane & 15;
  const int kb   = (lane >> 4) << 3;  // 0 or 8
  const bf16_t* p = Bm + (size_t)(col0 + n) * ld + k + kb;
  FragU f;
  f.u[0] = *(const uint4*)(p);        // K = k+kb .. k+kb+7
  f.u[1] = *(const uint4*)(p + 16);   // K = k+kb+16 .. k+kb+23
  return f.v;
}

__global__ __launch_bounds__(256) void cvt_f32_to_bf16(const float* __restrict__ in,
                                                       bf16_t* __restrict__ out,
                                                       int n) {
  int i = (blockIdx.x * 256 + threadIdx.x) * 4;
  if (i >= n) return;
  float4 f = *(const float4*)(in + i);
  union { bf16_t h[4]; uint2 u; } t;
  t.h[0] = (bf16_t)f.x; t.h[1] = (bf16_t)f.y;
  t.h[2] = (bf16_t)f.z; t.h[3] = (bf16_t)f.w;
  *(uint2*)(out + i) = t.u;
}

// C[m,n] = sum_k A[m,k] * W[n,k] + bias[n]  (torch Linear: x @ W^T + b)
// 8 waves/block, wave tile 64x64 (4x4 fragments), block tile 256(M) x 128(N).
// mode 0: bf16 out, [B,H,S,DK] ; mode 1: bf16 out, [B,H,DK,S] ; mode 2: f32 row-major.
__global__ __launch_bounds__(256) void gemm_bf16_wmma(const bf16_t* __restrict__ A,
                                                      const bf16_t* __restrict__ W,
                                                      const float* __restrict__ bias,
                                                      void* __restrict__ out,
                                                      int K, int N, int mode) {
  const int wid  = threadIdx.x >> 5;
  const int lane = threadIdx.x & 31;
  const int m0 = blockIdx.x * 256 + (wid & 3) * 64;
  const int n0 = blockIdx.y * 128 + (wid >> 2) * 64;

  v8f acc[4][4];
#pragma unroll
  for (int i = 0; i < 4; ++i)
#pragma unroll
    for (int j = 0; j < 4; ++j) acc[i][j] = v8f_zero();

  for (int k = 0; k < K; k += 32) {
    if (k + 32 < K) {  // uniform branch; speculative prefetch of next K-slab
      __builtin_prefetch((const void*)(A + (size_t)(m0 + lane) * K + k + 32), 0, 0);
      __builtin_prefetch((const void*)(W + (size_t)(n0 + lane) * K + k + 32), 0, 0);
    }
    v16bf af[4], bf[4];
#pragma unroll
    for (int i = 0; i < 4; ++i) af[i] = load_frag_a(A, m0 + 16 * i, k, K);
#pragma unroll
    for (int j = 0; j < 4; ++j) bf[j] = load_frag_b(W, n0 + 16 * j, k, K);
#pragma unroll
    for (int i = 0; i < 4; ++i)
#pragma unroll
      for (int j = 0; j < 4; ++j) acc[i][j] = wmma_acc(af[i], bf[j], acc[i][j]);
  }

  const int moff = (lane >> 4) << 3;  // C/D layout: vgpr j -> M = moff + j
  const int nl   = lane & 15;
#pragma unroll
  for (int i = 0; i < 4; ++i) {
#pragma unroll
    for (int j = 0; j < 4; ++j) {
#pragma unroll
      for (int r = 0; r < 8; ++r) {
        const int m = m0 + 16 * i + moff + r;
        const int n = n0 + 16 * j + nl;
        float vv = acc[i][j][r] + bias[n];
        if (mode == 2) {
          ((float*)out)[(size_t)m * N + n] = vv;
        } else {
          const int b = m >> 11, s = m & (SEQ - 1);
          const int h = n >> 6,  d = n & (DK - 1);
          bf16_t* o = (bf16_t*)out;
          if (mode == 0)
            o[(((size_t)(b * NHEAD + h) * SEQ + s) << 6) + d] = (bf16_t)vv;   // [B,H,S,DK]
          else
            o[(((size_t)(b * NHEAD + h) * DK + d) << 11) + s] = (bf16_t)vv;   // [B,H,DK,S]
        }
      }
    }
  }
}

// Flash attention: one wave per (b, h, 16-row q block). 32 keys per step.
__global__ __launch_bounds__(256) void attn_flash_wmma(const bf16_t* __restrict__ Qh,
                                                       const bf16_t* __restrict__ Kh,
                                                       const bf16_t* __restrict__ VhT,
                                                       const int* __restrict__ mask,
                                                       bf16_t* __restrict__ Xo) {
  __shared__ bf16_t pbuf[8][16 * 32];  // per-wave 16x32 bf16 P tile (layout shuffle)
  const int wid  = threadIdx.x >> 5;
  const int lane = threadIdx.x & 31;
  const int task = blockIdx.x * 8 + wid;
  const int q0   = (task & (SEQ / 16 - 1)) * 16;
  const int bh   = task >> 7;  // 0..63

  const bf16_t* Qp = Qh  + (size_t)bh * SEQ * DK;
  const bf16_t* Kp = Kh  + (size_t)bh * SEQ * DK;
  const bf16_t* Vp = VhT + (size_t)bh * DK * SEQ;

  const v16bf qf0 = load_frag_a(Qp, q0, 0, DK);   // Q rows, d = 0..31
  const v16bf qf1 = load_frag_a(Qp, q0, 32, DK);  // Q rows, d = 32..63

  v8f o[4];
#pragma unroll
  for (int nt = 0; nt < 4; ++nt) o[nt] = v8f_zero();
  float mrun[8], lrun[8];
#pragma unroll
  for (int r = 0; r < 8; ++r) { mrun[r] = -1e30f; lrun[r] = 0.0f; }

  const int   moff  = (lane >> 4) << 3;
  const int   nl    = lane & 15;
  const float scale = 0.125f;  // 1/sqrt(64)

  for (int s0 = 0; s0 < SEQ; s0 += 32) {
    // ---- scores: S = Q * K^T, two 16x16 tiles covering keys s0..s0+31 ----
    v8f sc[2];
#pragma unroll
    for (int t = 0; t < 2; ++t) {
      v16bf kf0 = load_frag_b(Kp, s0 + 16 * t, 0, DK);   // d 0..31
      v16bf kf1 = load_frag_b(Kp, s0 + 16 * t, 32, DK);  // d 32..63
      v8f z = v8f_zero();
      z = wmma_acc(qf0, kf0, z);
      z = wmma_acc(qf1, kf1, z);
      sc[t] = z;
    }
    // ---- scale + mask + per-row max (reduce across 16-lane halves) ----
    float rmax[8];
#pragma unroll
    for (int r = 0; r < 8; ++r) {
      const size_t mrow = (size_t)(q0 + moff + r) * SEQ + s0;
      float v0 = sc[0][r] * scale;
      float v1 = sc[1][r] * scale;
      if (mask[mrow + nl] == 0)      v0 = -1e9f;
      if (mask[mrow + 16 + nl] == 0) v1 = -1e9f;
      sc[0][r] = v0; sc[1][r] = v1;
      float mx = fmaxf(v0, v1);
#pragma unroll
      for (int off = 8; off; off >>= 1) mx = fmaxf(mx, __shfl_xor(mx, off, 32));
      rmax[r] = mx;
    }
    // ---- online softmax update ----
    float corr[8];
#pragma unroll
    for (int r = 0; r < 8; ++r) {
      const float mn = fmaxf(mrun[r], rmax[r]);
      corr[r] = __expf(mrun[r] - mn);
      mrun[r] = mn;
      const float p0 = __expf(sc[0][r] - mn);
      const float p1 = __expf(sc[1][r] - mn);
      sc[0][r] = p0; sc[1][r] = p1;
      float rs = p0 + p1;
#pragma unroll
      for (int off = 8; off; off >>= 1) rs += __shfl_xor(rs, off, 32);
      lrun[r] = lrun[r] * corr[r] + rs;
    }
#pragma unroll
    for (int nt = 0; nt < 4; ++nt)
#pragma unroll
      for (int r = 0; r < 8; ++r) o[nt][r] *= corr[r];

    // ---- P: C/D layout -> A layout via LDS ----
#pragma unroll
    for (int t = 0; t < 2; ++t)
#pragma unroll
      for (int r = 0; r < 8; ++r)
        pbuf[wid][(moff + r) * 32 + 16 * t + nl] = (bf16_t)sc[t][r];
    __syncthreads();
    FragU pf;
    {
      const int mm  = lane & 15;
      const int klo = (lane >> 4) << 3;
      pf.u[0] = *(const uint4*)&pbuf[wid][mm * 32 + klo];
      pf.u[1] = *(const uint4*)&pbuf[wid][mm * 32 + klo + 16];
    }
    // ---- O += P(16x32) * V(32x64) ----
#pragma unroll
    for (int nt = 0; nt < 4; ++nt) {
      v16bf vf = load_frag_b(Vp, 16 * nt, s0, SEQ);  // V^T: contiguous over s
      o[nt] = wmma_acc(pf.v, vf, o[nt]);
    }
    __syncthreads();
  }

  // ---- finalize: O /= l ; concat heads -> Xo[B,S,D] bf16 ----
  const int b = bh >> 4, h = bh & 15;
#pragma unroll
  for (int r = 0; r < 8; ++r) {
    const float rl = 1.0f / lrun[r];
    const size_t row = ((size_t)b * SEQ + q0 + moff + r) * D_MODEL + h * DK;
#pragma unroll
    for (int nt = 0; nt < 4; ++nt)
      Xo[row + 16 * nt + nl] = (bf16_t)(o[nt][r] * rl);
  }
}

extern "C" void kernel_launch(void* const* d_in, const int* in_sizes, int n_in,
                              void* d_out, int out_size, void* d_ws, size_t ws_size,
                              hipStream_t stream) {
  (void)in_sizes; (void)n_in; (void)out_size; (void)ws_size;
  const float* q   = (const float*)d_in[0];
  const float* k   = (const float*)d_in[1];
  const float* v   = (const float*)d_in[2];
  const int*   msk = (const int*)d_in[3];
  const float* w_q = (const float*)d_in[4];  const float* b_q = (const float*)d_in[5];
  const float* w_k = (const float*)d_in[6];  const float* b_k = (const float*)d_in[7];
  const float* w_v = (const float*)d_in[8];  const float* b_v = (const float*)d_in[9];
  const float* w_o = (const float*)d_in[10]; const float* b_o = (const float*)d_in[11];

  const size_t act = (size_t)MROWS * D_MODEL;    // 8Mi elements
  const size_t wts = (size_t)D_MODEL * D_MODEL;  // 1Mi elements

  char* p = (char*)d_ws;
  bf16_t* qb  = (bf16_t*)p; p += act * 2;
  bf16_t* kb  = (bf16_t*)p; p += act * 2;
  bf16_t* vb  = (bf16_t*)p; p += act * 2;
  bf16_t* wqb = (bf16_t*)p; p += wts * 2;
  bf16_t* wkb = (bf16_t*)p; p += wts * 2;
  bf16_t* wvb = (bf16_t*)p; p += wts * 2;
  bf16_t* wob = (bf16_t*)p; p += wts * 2;
  bf16_t* Qh  = (bf16_t*)p; p += act * 2;   // [B,H,S,DK]
  bf16_t* Kh  = (bf16_t*)p; p += act * 2;   // [B,H,S,DK]
  bf16_t* VhT = (bf16_t*)p; p += act * 2;   // [B,H,DK,S]
  bf16_t* Xa  = (bf16_t*)p; p += act * 2;   // [B,S,D] bf16 (pre output-proj)

  const int actBlocks = (int)(act / 4 / 256);  // 8192
  const int wtsBlocks = (int)(wts / 4 / 256);  // 1024
  cvt_f32_to_bf16<<<actBlocks, 256, 0, stream>>>(q, qb, (int)act);
  cvt_f32_to_bf16<<<actBlocks, 256, 0, stream>>>(k, kb, (int)act);
  cvt_f32_to_bf16<<<actBlocks, 256, 0, stream>>>(v, vb, (int)act);
  cvt_f32_to_bf16<<<wtsBlocks, 256, 0, stream>>>(w_q, wqb, (int)wts);
  cvt_f32_to_bf16<<<wtsBlocks, 256, 0, stream>>>(w_k, wkb, (int)wts);
  cvt_f32_to_bf16<<<wtsBlocks, 256, 0, stream>>>(w_v, wvb, (int)wts);
  cvt_f32_to_bf16<<<wtsBlocks, 256, 0, stream>>>(w_o, wob, (int)wts);

  dim3 gg(MROWS / 256, D_MODEL / 128);
  gemm_bf16_wmma<<<gg, 256, 0, stream>>>(qb, wqb, b_q, Qh, D_MODEL, D_MODEL, 0);
  gemm_bf16_wmma<<<gg, 256, 0, stream>>>(kb, wkb, b_k, Kh, D_MODEL, D_MODEL, 0);
  gemm_bf16_wmma<<<gg, 256, 0, stream>>>(vb, wvb, b_v, VhT, D_MODEL, D_MODEL, 1);

  attn_flash_wmma<<<(BATCH * NHEAD * (SEQ / 16)) / 8, 256, 0, stream>>>(Qh, Kh, VhT,
                                                                        msk, Xa);

  gemm_bf16_wmma<<<gg, 256, 0, stream>>>(Xa, wob, b_o, d_out, D_MODEL, D_MODEL, 2);
}
